// DH_SNN_Network_86474871538232
// MI455X (gfx1250) — matrix operand order repaired and checked by reading, not compiled, not run on hardware
//
#include <hip/hip_runtime.h>
#include <cstddef>

#define B_    128
#define T_    250
#define IN_   700
#define KPAD  704
#define KT_   22      // KPAD/32
#define H_    512
#define NB_   4
#define N_    2048    // H_*NB_
#define OUT_  20
#define VTH   0.5f
#define HW_   64      // h per workgroup (4 waves x 16 h)
#define NHW   (H_/HW_)   // 8 workgroups along h
#define NBT   (B_/16)    // 8 batch tiles
#define NSLICE 32        // NHW*4 per-wave partial slices per batch tile

typedef __attribute__((ext_vector_type(16))) __bf16 v16bf;
typedef __attribute__((ext_vector_type(8)))  __bf16 v8bf;
typedef __attribute__((ext_vector_type(8)))  float  v8f;

__device__ __forceinline__ unsigned short f2bf(float f) {
  union { float f; unsigned u; } v; v.f = f;
  unsigned r = v.u + 0x7fffu + ((v.u >> 16) & 1u);   // round-to-nearest-even
  return (unsigned short)(r >> 16);
}
__device__ __forceinline__ float sigm(float x) { return 1.0f / (1.0f + __expf(-x)); }

// ---------- Phase 0a: masked W1 -> bf16, K padded, rows permuted BRANCH-MAJOR:
// wm[(j*H + h)][k] = W1[(h*NB+j)][k] * mask[...]   (j = branch, h = neuron)
__global__ void prep_wm(const float* __restrict__ W1, const float* __restrict__ mask,
                        unsigned short* __restrict__ wm) {
  int i = blockIdx.x * 256 + threadIdx.x;
  if (i >= N_ * KPAD) return;
  int np = i / KPAD, k = i - np * KPAD;   // np = j*H + h
  int j = np / H_, h = np - j * H_;
  int n = h * NB_ + j;                    // original row
  float v = 0.0f;
  if (k < IN_) { int idx = n * IN_ + k; v = W1[idx] * mask[idx]; }
  wm[i] = f2bf(v);
}

// ---------- Phase 0b: x -> bf16, K padded, row-major [(b*T+t)][KPAD]
__global__ void prep_x(const float* __restrict__ x, unsigned short* __restrict__ xb) {
  long long i = (long long)blockIdx.x * 256 + threadIdx.x;
  if (i >= (long long)B_ * T_ * KPAD) return;
  long long row = i / KPAD; int k = (int)(i - row * KPAD);
  float v = (k < IN_) ? x[row * IN_ + k] : 0.0f;
  xb[i] = f2bf(v);
}

// ---------- Phase 1: fused WMMA GEMM + temporal scan, barrier-free main loop.
// Wave owns 16 h (all 4 branches => 4 N-tiles sharing one A fragment).
// WMMA C/D layout: lane holds N=lane&15, vgpr r -> M = r + 8*(lane>>4).
__global__ __launch_bounds__(128) void snn_scan(
    const unsigned short* __restrict__ xb,     // bf16 X
    const unsigned short* __restrict__ wm,     // bf16 branch-major masked W1
    const float* __restrict__ b1,
    const float* __restrict__ tau_n,
    const float* __restrict__ tau_m_h,
    const float* __restrict__ W2,
    const float* __restrict__ tau_m_ro,
    float* __restrict__ Pbuf, float* __restrict__ Gbuf) {
  const int tid  = threadIdx.x;
  const int wave = tid >> 5;
  const int lane = tid & 31;
  const int half = lane >> 4;
  const int lm   = lane & 15;

  const int bt  = blockIdx.y;            // batch tile (16 rows)
  const int hwg = blockIdx.x;            // h tile of 64
  const int hw0 = hwg * HW_ + wave * 16; // wave's first h
  const int h   = hw0 + lm;              // this lane's neuron

  __shared__ float w2s[OUT_][HW_];       // W2 tile for workgroup's 64 h
  __shared__ float sM[4][16][17];        // per-wave spike tiles [b][h_local]

  for (int p = tid; p < OUT_ * HW_; p += 128)
    w2s[p / HW_][p % HW_] = W2[(p / HW_) * H_ + hwg * HW_ + (p % HW_)];

  const float alpha = sigm(tau_m_h[0]);
  float beta[NB_], omb[NB_], b1v[NB_];
  #pragma unroll
  for (int j = 0; j < NB_; ++j) {
    beta[j] = sigm(tau_n[j]);            // wave-uniform -> scalar regs
    omb[j]  = 1.0f - beta[j];
    b1v[j]  = b1[h * NB_ + j];
  }

  float d[NB_][8], vv[8], ss[8];
  #pragma unroll
  for (int r = 0; r < 8; ++r) {
    vv[r] = 0.f; ss[r] = 0.f;
    #pragma unroll
    for (int j = 0; j < NB_; ++j) d[j][r] = 0.f;
  }

  // per-thread readout partials: pairs p = lane + 32*jj (exactly 320 = 32*10)
  float Pr[10], Gr[10], gamr[10];
  #pragma unroll
  for (int jj = 0; jj < 10; ++jj) {
    int p = lane + 32 * jj;
    Pr[jj] = 0.f; Gr[jj] = 0.f;
    gamr[jj] = sigm(tau_m_ro[p % OUT_]);
  }

  // A fragment (16-bit 16x32): lane holds M=lm; K = 16*half + 0..15 contiguous
  const unsigned short* xrow0 = xb + (size_t)(bt * 16 + lm) * T_ * KPAD + half * 16;
  // B fragments (16-bit 32x16): lane holds N=lm; K = 16*half + 0..15 contiguous.
  // branch j's row lives at +j*H_*KPAD (folds into the 24-bit inst offset).
  const unsigned short* wrow = wm + (size_t)h * KPAD + half * 16;

  __syncthreads();                       // w2s ready; only barrier in kernel

  for (int t = 0; t < T_; ++t) {
    const unsigned short* xr = xrow0 + (size_t)t * KPAD;
    v8f c[NB_];
    #pragma unroll
    for (int j = 0; j < NB_; ++j)
      #pragma unroll
      for (int r = 0; r < 8; ++r) c[j][r] = 0.f;

    #pragma unroll 2
    for (int kt = 0; kt < KT_; ++kt) {
      const int kb = kt * 32;
      union { v16bf v; v8bf hh[2]; } A;
      A.hh[0] = *(const v8bf*)(xr + kb);        // K = kb + 16*half + 0..7
      A.hh[1] = *(const v8bf*)(xr + kb + 8);    // K = kb + 16*half + 8..15
      #pragma unroll
      for (int j = 0; j < NB_; ++j) {
        const v16bf Bm = *(const v16bf*)(wrow + (size_t)j * H_ * KPAD + kb);
        c[j] = __builtin_amdgcn_wmma_f32_16x16x32_bf16(false, A.v, false, Bm,
                                                       (short)0, c[j], false, false);
      }
    }

    // state update: all 4 branch currents of (b,h) live in this lane
    #pragma unroll
    for (int r = 0; r < 8; ++r) {
      float som = 0.f;
      #pragma unroll
      for (int j = 0; j < NB_; ++j) {
        float di = c[j][r] + b1v[j];
        d[j][r] = beta[j] * d[j][r] + omb[j] * di;
        som += d[j][r];
      }
      vv[r] = vv[r] * alpha + (1.f - alpha) * som - VTH * ss[r];
      ss[r] = (vv[r] >= VTH) ? 1.f : 0.f;
    }

    // wave-private spike tile (same-wave LDS is in-order; no s_barrier)
    __builtin_amdgcn_wave_barrier();
    #pragma unroll
    for (int r = 0; r < 8; ++r) sM[wave][half * 8 + r][lm] = ss[r];
    __builtin_amdgcn_wave_barrier();

    // per-wave readout partial: y[b,o] over this wave's 16 h
    #pragma unroll
    for (int jj = 0; jj < 10; ++jj) {
      int p = lane + 32 * jj;
      int b = p / OUT_, o = p - b * OUT_;
      float y = 0.f;
      #pragma unroll
      for (int hl = 0; hl < 16; ++hl)
        y += w2s[o][wave * 16 + hl] * sM[wave][b][hl];
      Pr[jj] += y;
      Gr[jj] = gamr[jj] * (Gr[jj] + y);
    }
    __builtin_amdgcn_wave_barrier();
  }

  // slice = bt*32 + hwg*4 + wave; 320 floats per slice
  const size_t base = (((size_t)bt * NHW + hwg) * 4 + wave) * (16 * OUT_);
  #pragma unroll
  for (int jj = 0; jj < 10; ++jj) {
    int p = lane + 32 * jj;
    Pbuf[base + p] = Pr[jj];
    Gbuf[base + p] = Gr[jj];
  }
}

// ---------- Phase 2: reduce the 32 per-wave slices per batch tile
// out[b,o] = (P - G)/T + b2[o]*(1 - (1/T)*sum_{k=1..T} gam^k)
__global__ void readout(const float* __restrict__ Pbuf, const float* __restrict__ Gbuf,
                        const float* __restrict__ b2, const float* __restrict__ tau_m_ro,
                        float* __restrict__ out) {
  int i = blockIdx.x * 256 + threadIdx.x;
  if (i >= B_ * OUT_) return;
  int b = i / OUT_, o = i - b * OUT_;
  int bt = b >> 4, bl = b & 15;
  float sp = 0.f, sg = 0.f;
  for (int s = 0; s < NSLICE; ++s) {
    size_t base = ((size_t)bt * NSLICE + s) * (16 * OUT_) + bl * OUT_ + o;
    sp += Pbuf[base];
    sg += Gbuf[base];
  }
  float gam = sigm(tau_m_ro[o]);
  float g = 1.f, acc = 0.f;
  for (int k = 0; k < T_; ++k) { g *= gam; acc += g; }
  float coef = 1.f - acc / (float)T_;
  out[i] = (sp - sg) / (float)T_ + b2[o] * coef;
}

extern "C" void kernel_launch(void* const* d_in, const int* in_sizes, int n_in,
                              void* d_out, int out_size, void* d_ws, size_t ws_size,
                              hipStream_t stream) {
  const float* x        = (const float*)d_in[0];
  const float* W1       = (const float*)d_in[1];
  const float* b1       = (const float*)d_in[2];
  const float* tau_n    = (const float*)d_in[3];
  const float* tau_m_h  = (const float*)d_in[4];
  const float* W2       = (const float*)d_in[5];
  const float* b2       = (const float*)d_in[6];
  const float* tau_m_ro = (const float*)d_in[7];
  const float* mask     = (const float*)d_in[8];
  float* out = (float*)d_out;

  // workspace layout
  unsigned short* xb = (unsigned short*)d_ws;                      // 45.06 MB
  size_t xbBytes = (size_t)B_ * T_ * KPAD * sizeof(unsigned short);
  unsigned short* wm = (unsigned short*)((char*)d_ws + xbBytes);   // 2.88 MB
  size_t wmBytes = (size_t)N_ * KPAD * sizeof(unsigned short);
  float* Pbuf = (float*)((char*)d_ws + xbBytes + wmBytes);         // 0.33 MB
  float* Gbuf = Pbuf + (size_t)NBT * NSLICE * 16 * OUT_;           // 0.33 MB

  {
    long long nx = (long long)B_ * T_ * KPAD;
    prep_x<<<(unsigned)((nx + 255) / 256), 256, 0, stream>>>(x, xb);
  }
  prep_wm<<<(N_ * KPAD + 255) / 256, 256, 0, stream>>>(W1, mask, wm);

  dim3 grid(NHW, NBT);   // 8 x 8 workgroups, 4 waves each
  snn_scan<<<grid, 128, 0, stream>>>(xb, wm, b1, tau_n, tau_m_h, W2, tau_m_ro,
                                     Pbuf, Gbuf);

  readout<<<(B_ * OUT_ + 255) / 256, 256, 0, stream>>>(Pbuf, Gbuf, b2, tau_m_ro, out);
}
